// MultiheadFlashDiff_69286412419251
// MI455X (gfx1250) — matrix operand order repaired and checked by reading, not compiled, not run on hardware
//
#include <hip/hip_runtime.h>
#include <cstdint>

typedef __attribute__((ext_vector_type(16))) _Float16 v16h;
typedef __attribute__((ext_vector_type(8)))  float    v8f;
typedef __attribute__((ext_vector_type(4)))  unsigned int v4u;
typedef __attribute__((ext_vector_type(8)))  int      v8i;
typedef __attribute__((ext_vector_type(4)))  int      v4i;

#define DEV __device__ __forceinline__

static constexpr int   Sdim = 2048;
static constexpr int   Edim = 2048;
static constexpr int   Hn   = 16;
static constexpr int   HDd  = 64;   // per-half head dim
static constexpr int   DVd  = 128;  // v head dim
static constexpr float LAMBDA_INIT = 0.783605766531622f; // 0.8 - 0.6*exp(-3.6)
static constexpr float GAIN = 1.0f - LAMBDA_INIT;
static constexpr float NEGF = -1e30f;
static constexpr float EPSF = 1e-5f;

// ---------------------------------------------------------------------------
// WMMA helpers: f16 16x16x32, f32 accumulate.
// A layout (16x32, MxK): lane l (lr=l&15,hi=l>>4) holds row lr;
//   dword vg holds K = 16*(vg>>2) + 8*hi + 2*(vg&3) + {0,1}
// B layout (32x16, KxN): lane holds col lr; dword vg holds K = 16*(vg>>2) + 8*hi + 2*(vg&3) + {0,1}
// C layout (16x16): vgpr r, lanes 0-15 -> M=r, lanes 16-31 -> M=r+8; N=lr
// ---------------------------------------------------------------------------
DEV v8f wmma16(v16h a, v16h b, v8f c) {
  return __builtin_amdgcn_wmma_f32_16x16x32_f16(false, a, false, b, (short)0, c,
                                                false, false);
}

DEV v16h frag_a(const _Float16* tile, int ld, int row, int hi, int kc) {
  union { v16h v; uint32_t u[8]; } f;
  const _Float16* rp = tile + row * ld + kc + 8 * hi;
#pragma unroll
  for (int vg = 0; vg < 8; ++vg) {
    int k0 = ((vg >> 2) << 4) + ((vg & 3) << 1);
    f.u[vg] = *(const uint32_t*)(rp + k0);
  }
  return f.v;
}

// tileT is the B matrix stored [col][k] so K is contiguous.
DEV v16h frag_b(const _Float16* tileT, int ld, int col, int hi, int kc) {
  union { v16h v; uint32_t u[8]; } f;
  const _Float16* cp = tileT + col * ld + kc + 8 * hi;
#pragma unroll
  for (int vg = 0; vg < 8; ++vg) f.u[vg] = *(const uint32_t*)(cp + ((vg >> 2) << 4) + ((vg & 3) << 1));
  return f.v;
}

// ---------------------------------------------------------------------------
// CDNA5 async global->LDS copy (ASYNCcnt) — per-lane 16B, bypasses VGPRs.
// LDS byte offset = low 32 bits of the flat shared-pointer (aperture rule).
// ---------------------------------------------------------------------------
DEV void async_ld_b128(void* lds_dst, const void* gsrc) {
  uint32_t loff = (uint32_t)(uintptr_t)lds_dst;
  asm volatile("global_load_async_to_lds_b128 %0, %1, off"
               :: "v"(loff), "v"(gsrc) : "memory");
}
DEV void wait_async0() { asm volatile("s_wait_asynccnt 0x0" ::: "memory"); }

// ---------------------------------------------------------------------------
// fp32 -> f16 conversion
// ---------------------------------------------------------------------------
__global__ void cvt_f16(const float* __restrict__ in, _Float16* __restrict__ out,
                        int n) {
  int i = blockIdx.x * blockDim.x + threadIdx.x;
  if (i < n) out[i] = (_Float16)in[i];
}

// ---------------------------------------------------------------------------
// 2048x2048x2048 GEMM: C = A * B, A row-major [M][K] f16, B row-major [K][N] f16
// Block tile 64x64, BK=32, 128 threads = 4 waves, each wave 2x2 16x16 tiles.
// A tile staged by the Tensor Data Mover (TDM) with descriptor-level LDS row
// padding (32 halves + 4 pad -> stride 36); B staged via VGPR transpose.
// ---------------------------------------------------------------------------
template <typename OutT>
__global__ __launch_bounds__(128) void gemm2048(const _Float16* __restrict__ A,
                                                const _Float16* __restrict__ Bw,
                                                OutT* __restrict__ C) {
  const int N = Edim, K = Edim;
  __shared__ __align__(16) _Float16 As[64][36];  // [m][k] (TDM-padded rows)
  __shared__ __align__(16) _Float16 Bt[64][36];  // [n][k] (transposed)
  const int tid = threadIdx.x;
  const int lane = tid & 31, wave = tid >> 5;
  const int hi = lane >> 4, lr = lane & 15;
  const int wm = wave >> 1, wn = wave & 1;
  const int rowBase = blockIdx.y * 64, colBase = blockIdx.x * 64;

  v8f acc[2][2] = {};
  const uint32_t lds0 = (uint32_t)(uintptr_t)&As[0][0];

  for (int k0 = 0; k0 < K; k0 += 32) {
    if (wave == 0) {
      // ---- TDM descriptor: 2D tile 32(x) x 64(y) halves from [2048][2048] ----
      uint64_t ga = (uint64_t)(uintptr_t)&A[(size_t)rowBase * K + k0];
      v4u g0 = { 1u,                                   // count=1, user mode
                 lds0,                                 // lds_addr
                 (uint32_t)ga,                         // global_addr[31:0]
                 ((uint32_t)(ga >> 32) & 0x01FFFFFFu) | (2u << 30) }; // type=2
      v8i g1 = { (int)((1u << 16)     // data_size = 2 bytes
                     | (1u << 20)     // pad_enable
                     | (3u << 22)     // pad_interval: 16 dwords (=32 halves)
                     | (1u << 25)),   // pad_amount: 2 dwords (=4 halves)
                 (int)(2048u << 16),  // tensor_dim0 lo16 (atomic_addr=0)
                 (int)(2048u << 16),  // tensor_dim0 hi=0 | tensor_dim1 lo16
                 (int)(32u << 16),    // tensor_dim1 hi=0 | tile_dim0=32
                 64,                  // tile_dim1=64, tile_dim2=0
                 2048,                // tensor_dim0_stride lo32
                 0, 0 };              // stride hi / dim1_stride (unused, 2D)
      v4i g2 = { 0, 0, 0, 0 };
      v4i g3 = { 0, 0, 0, 0 };
      v8i g4 = { 0, 0, 0, 0, 0, 0, 0, 0 };
      __builtin_amdgcn_tensor_load_to_lds(g0, g1, g2, g3, g4, 0);
    }
#pragma unroll
    for (int i = 0; i < 8; ++i) {  // B tile: 32x64 halves, transposed into Bt
      int idx = tid + i * 128;
      int kr = idx >> 5, c2 = (idx & 31) << 1;
      union { uint32_t u; _Float16 h[2]; } t;
      t.u = *(const uint32_t*)&Bw[(size_t)(k0 + kr) * N + colBase + c2];
      Bt[c2][kr] = t.h[0];
      Bt[c2 + 1][kr] = t.h[1];
    }
    if (wave == 0) __builtin_amdgcn_s_wait_tensorcnt(0);
    __syncthreads();
#pragma unroll
    for (int mi = 0; mi < 2; ++mi) {
      v16h af = frag_a(&As[wm * 32 + mi * 16][0], 36, lr, hi, 0);
#pragma unroll
      for (int ni = 0; ni < 2; ++ni) {
        v16h bf = frag_b(&Bt[wn * 32 + ni * 16][0], 36, lr, hi, 0);
        acc[mi][ni] = wmma16(af, bf, acc[mi][ni]);
      }
    }
    __syncthreads();
  }
#pragma unroll
  for (int mi = 0; mi < 2; ++mi)
#pragma unroll
    for (int ni = 0; ni < 2; ++ni) {
      int row0 = rowBase + wm * 32 + mi * 16 + 8 * hi;
      int col = colBase + wn * 32 + ni * 16 + lr;
#pragma unroll
      for (int r = 0; r < 8; ++r)
        C[(size_t)(row0 + r) * N + col] = (OutT)acc[mi][ni][r];
    }
}

// ---------------------------------------------------------------------------
// In-place interleaved RoPE on [S][2H][HD] f16 (pairs (2i,2i+1) within HD).
// ---------------------------------------------------------------------------
__global__ void rope_ip(_Float16* __restrict__ qk) {
  int p = blockIdx.x * blockDim.x + threadIdx.x;
  const int NP = Sdim * 2 * Hn * (HDd / 2);
  if (p >= NP) return;
  int i = p & 31;          // rotary frequency index (HD/2 = 32)
  int hh = (p >> 5) & 31;  // head-half index (2H = 32)
  int s = p >> 10;         // sequence position
  const float LOG10000 = 9.210340371976184f;
  float div = __expf(-(LOG10000 / (float)HDd) * (2.0f * (float)i));
  float ang = (float)s * div;
  float c = __cosf(ang), sn = __sinf(ang);
  _Float16* base = qk + ((size_t)s * 32 + hh) * HDd + 2 * i;
  float x1 = (float)base[0], x2 = (float)base[1];
  base[0] = (_Float16)(x1 * c - x2 * sn);
  base[1] = (_Float16)(x1 * sn + x2 * c);
}

// ---------------------------------------------------------------------------
// lambda_full = exp(sum lq1*lk1) - exp(sum lq2*lk2) + LAMBDA_INIT
// ---------------------------------------------------------------------------
__global__ void lambda_k(const float* __restrict__ lq1, const float* __restrict__ lk1,
                         const float* __restrict__ lq2, const float* __restrict__ lk2,
                         float* __restrict__ lam) {
  __shared__ float s1[64], s2[64];
  int t = threadIdx.x;
  s1[t] = lq1[t] * lk1[t];
  s2[t] = lq2[t] * lk2[t];
  __syncthreads();
  if (t == 0) {
    float a = 0.f, b = 0.f;
    for (int i = 0; i < 64; ++i) { a += s1[i]; b += s2[i]; }
    *lam = __expf(a) - __expf(b) + LAMBDA_INIT;
  }
}

// ---------------------------------------------------------------------------
// Flash differential attention + diff-combine + RMSNorm, one head per blockIdx.y,
// 64 queries per blockIdx.x, 128 threads (4 waves, 16 query rows each).
// ---------------------------------------------------------------------------
DEV void online_update(v8f sa, v8f sb, float* m, float* l, v8f* o,
                       _Float16 (*P)[36], int lr, int hi) {
#pragma unroll
  for (int r = 0; r < 8; ++r) {
    float ea = sa[r], eb = sb[r];
    float mx = fmaxf(ea, eb);
#pragma unroll
    for (int off = 1; off < 16; off <<= 1) mx = fmaxf(mx, __shfl_xor(mx, off, 32));
    float mnew = fmaxf(m[r], mx);
    float pa = (ea < -1e29f) ? 0.0f : __expf(ea - mnew);
    float pb = (eb < -1e29f) ? 0.0f : __expf(eb - mnew);
    float sum = pa + pb;
#pragma unroll
    for (int off = 1; off < 16; off <<= 1) sum += __shfl_xor(sum, off, 32);
    float alpha = __expf(m[r] - mnew);
    m[r] = mnew;
    l[r] = l[r] * alpha + sum;
#pragma unroll
    for (int t = 0; t < 8; ++t) o[t][r] *= alpha;
    int prow = r + 8 * hi;
    P[prow][lr] = (_Float16)pa;
    P[prow][16 + lr] = (_Float16)pb;
  }
}

__global__ __launch_bounds__(128) void flashdiff(
    const _Float16* __restrict__ qh, const _Float16* __restrict__ kh,
    const _Float16* __restrict__ vh, const float* __restrict__ g,
    const float* __restrict__ lamp, _Float16* __restrict__ ah) {
  const int h = blockIdx.y;
  const int s0 = blockIdx.x * 64;
  const int tid = threadIdx.x;
  const int lane = tid & 31, wave = tid >> 5;
  const int hi = lane >> 4, lr = lane & 15;
  const int qr = wave * 16;

  __shared__ __align__(16) _Float16 Q1[64][68];
  __shared__ __align__(16) _Float16 Q2[64][68];
  __shared__ __align__(16) _Float16 K1[32][68];
  __shared__ __align__(16) _Float16 K2[32][68];
  __shared__ __align__(16) _Float16 Vt[128][36];   // [dv][key]
  __shared__ __align__(16) _Float16 P1[4][16][36]; // per-wave P tiles
  __shared__ __align__(16) _Float16 P2[4][16][36];

  // stage Q tiles (64x64 per variant) via async global->LDS b128 copies
#pragma unroll
  for (int i = 0; i < 4; ++i) {
    int idx = tid + i * 128;
    int r = idx >> 3, c = (idx & 7) << 3;  // c in halves (8 halves = 16B)
    async_ld_b128(&Q1[r][c], &qh[((size_t)(s0 + r) * 32 + 2 * h) * HDd + c]);
    async_ld_b128(&Q2[r][c], &qh[((size_t)(s0 + r) * 32 + 2 * h + 1) * HDd + c]);
  }
  wait_async0();
  __syncthreads();
  v16h qf1[2], qf2[2];
#pragma unroll
  for (int kc = 0; kc < 2; ++kc) {
    qf1[kc] = frag_a(&Q1[qr][0], 68, lr, hi, kc * 32);
    qf2[kc] = frag_a(&Q2[qr][0], 68, lr, hi, kc * 32);
  }

  v8f o1[8] = {}, o2[8] = {};
  float m1[8], m2[8], l1[8], l2[8];
#pragma unroll
  for (int r = 0; r < 8; ++r) { m1[r] = m2[r] = NEGF; l1[r] = l2[r] = 0.f; }
  const float scale = 0.125f;  // 1/sqrt(64)

  const int nkt = 2 * (blockIdx.x + 1);  // causal: keys up to s0+63
  for (int kt = 0; kt < nkt; ++kt) {
    const int kb = kt * 32;
    __syncthreads();
#pragma unroll
    for (int i = 0; i < 2; ++i) {  // K tiles 32x64 each, async copies
      int idx = tid + i * 128;
      int r = idx >> 3, c = (idx & 7) << 3;
      async_ld_b128(&K1[r][c], &kh[((size_t)(kb + r) * 32 + 2 * h) * HDd + c]);
      async_ld_b128(&K2[r][c], &kh[((size_t)(kb + r) * 32 + 2 * h + 1) * HDd + c]);
    }
    for (int i = tid; i < 1024; i += 128) {  // V 16x128 (keys kb..kb+15), transposed into Vt
      int r = i >> 6, c2 = (i & 63) << 1;
      union { uint32_t u; _Float16 hh[2]; } t;
      t.u = *(const uint32_t*)&vh[((size_t)(kb + r) * Hn + h) * DVd + c2];
      Vt[c2][r] = t.hh[0];
      Vt[c2 + 1][r] = t.hh[1];
    }
    wait_async0();
    __syncthreads();

    // scores: 16 queries x 32 keys per wave, both variants
    v8f s1a = {}, s1b = {}, s2a = {}, s2b = {};
#pragma unroll
    for (int kc = 0; kc < 2; ++kc) {
      v16h b1a = frag_b(&K1[0][0],  68, lr, hi, kc * 32);
      v16h b1b = frag_b(&K1[16][0], 68, lr, hi, kc * 32);
      v16h b2a = frag_b(&K2[0][0],  68, lr, hi, kc * 32);
      v16h b2b = frag_b(&K2[16][0], 68, lr, hi, kc * 32);
      s1a = wmma16(qf1[kc], b1a, s1a);
      s1b = wmma16(qf1[kc], b1b, s1b);
      s2a = wmma16(qf2[kc], b2a, s2a);
      s2b = wmma16(qf2[kc], b2b, s2b);
    }
    // scale + causal mask
    const int col_a = kb + lr, col_b = kb + 16 + lr;
#pragma unroll
    for (int r = 0; r < 8; ++r) {
      int row = s0 + qr + r + 8 * hi;
      s1a[r] = (col_a > row) ? NEGF : s1a[r] * scale;
      s1b[r] = (col_b > row) ? NEGF : s1b[r] * scale;
      s2a[r] = (col_a > row) ? NEGF : s2a[r] * scale;
      s2b[r] = (col_b > row) ? NEGF : s2b[r] * scale;
    }
    online_update(s1a, s1b, m1, l1, o1, P1[wave], lr, hi);
    online_update(s2a, s2b, m2, l2, o2, P2[wave], lr, hi);

    // P * V  (per-wave LDS; same-wave DS ops are in-order)
    v16h pf1 = frag_a(&P1[wave][0][0], 36, lr, hi, 0);
    v16h pf2 = frag_a(&P2[wave][0][0], 36, lr, hi, 0);
#pragma unroll
    for (int t = 0; t < 8; ++t) {
      v16h vf = frag_b(&Vt[t * 16][0], 36, lr, hi, 0);
      o1[t] = wmma16(pf1, vf, o1[t]);
      o2[t] = wmma16(pf2, vf, o2[t]);
    }
  }

  // epilogue: diff-combine, RMSNorm over 128, g, gain
  const float lam = lamp[0];
  float gv[8];
#pragma unroll
  for (int t = 0; t < 8; ++t) gv[t] = g[t * 16 + lr];
#pragma unroll
  for (int r = 0; r < 8; ++r) {
    float inv1 = 1.0f / l1[r];
    float inv2 = 1.0f / l2[r];
    float av[8], ss = 0.f;
#pragma unroll
    for (int t = 0; t < 8; ++t) {
      float a = o1[t][r] * inv1 - lam * (o2[t][r] * inv2);
      av[t] = a;
      ss += a * a;
    }
#pragma unroll
    for (int off = 1; off < 16; off <<= 1) ss += __shfl_xor(ss, off, 32);
    float rs = rsqrtf(ss * (1.0f / 128.0f) + EPSF);
    int row = s0 + qr + r + 8 * hi;
#pragma unroll
    for (int t = 0; t < 8; ++t) {
      float outv = av[t] * rs * gv[t] * GAIN;
      ah[(size_t)row * Edim + h * DVd + t * 16 + lr] = (_Float16)outv;
    }
  }
}

// ---------------------------------------------------------------------------
extern "C" void kernel_launch(void* const* d_in, const int* in_sizes, int n_in,
                              void* d_out, int out_size, void* d_ws, size_t ws_size,
                              hipStream_t stream) {
  (void)in_sizes; (void)n_in; (void)out_size; (void)ws_size;
  const float* x   = (const float*)d_in[0];
  const float* Wq  = (const float*)d_in[1];
  const float* Wk  = (const float*)d_in[2];
  const float* Wv  = (const float*)d_in[3];
  const float* Wo  = (const float*)d_in[4];
  const float* lq1 = (const float*)d_in[5];
  const float* lk1 = (const float*)d_in[6];
  const float* lq2 = (const float*)d_in[7];
  const float* lk2 = (const float*)d_in[8];
  const float* g   = (const float*)d_in[9];

  const size_t SZ = (size_t)Edim * Edim;  // 4M elements
  _Float16* xh  = (_Float16*)d_ws;
  _Float16* Wqh = xh + SZ;
  _Float16* Wkh = Wqh + SZ;
  _Float16* Wvh = Wkh + SZ;
  _Float16* Woh = Wvh + SZ;
  _Float16* qhp = Woh + SZ;
  _Float16* khp = qhp + SZ;
  _Float16* vhp = khp + SZ;
  _Float16* ahp = vhp + SZ;
  float* lamw = (float*)(ahp + SZ);

  const int n = (int)SZ;
  dim3 cb(256), cg((n + 255) / 256);
  cvt_f16<<<cg, cb, 0, stream>>>(x, xh, n);
  cvt_f16<<<cg, cb, 0, stream>>>(Wq, Wqh, n);
  cvt_f16<<<cg, cb, 0, stream>>>(Wk, Wkh, n);
  cvt_f16<<<cg, cb, 0, stream>>>(Wv, Wvh, n);
  cvt_f16<<<cg, cb, 0, stream>>>(Wo, Woh, n);

  dim3 gb(128), gg(Edim / 64, Sdim / 64);
  gemm2048<_Float16><<<gg, gb, 0, stream>>>(xh, Wqh, qhp);
  gemm2048<_Float16><<<gg, gb, 0, stream>>>(xh, Wkh, khp);
  gemm2048<_Float16><<<gg, gb, 0, stream>>>(xh, Wvh, vhp);

  const int np = Sdim * 2 * Hn * (HDd / 2);
  rope_ip<<<(np + 255) / 256, 256, 0, stream>>>(qhp);
  rope_ip<<<(np + 255) / 256, 256, 0, stream>>>(khp);

  lambda_k<<<1, 64, 0, stream>>>(lq1, lk1, lq2, lk2, lamw);

  flashdiff<<<dim3(Sdim / 64, Hn), 128, 0, stream>>>(qhp, khp, vhp, g, lamw, ahp);

  gemm2048<float><<<gg, gb, 0, stream>>>(ahp, Woh, (float*)d_out);
}
